// MixModel_13769665151544
// MI455X (gfx1250) — matrix operand order repaired and checked by reading, not compiled, not run on hardware
//
#include <hip/hip_runtime.h>
#include <hip/hip_bf16.h>

typedef __attribute__((ext_vector_type(16))) __bf16 v16bf;
typedef __attribute__((ext_vector_type(8)))  float  v8f;

#define NNODES 100000
#define NEDGES 1250000

#define WMMA_BF16(a,b,c) __builtin_amdgcn_wmma_f32_16x16x32_bf16(false,(a),false,(b),(short)0,(c),false,false)

// ---- WMMA fragment helpers (CDNA5 16x16x32 bf16 layouts, wave32) ----
// A 16x32 (MxK): lane half h, row m=lane&15; elems 0..7 -> K=k0+8h+i,
//                elems 8..15 -> K=k0+16+8h+(i-8)  (two contiguous 8-runs)
struct f16run { float v[16]; };
__device__ __forceinline__ f16run load_run16(const float* p) {
  f16run r;
  float4 x0 = *(const float4*)(p);
  float4 x1 = *(const float4*)(p + 4);
  float4 x2 = *(const float4*)(p + 16);
  float4 x3 = *(const float4*)(p + 20);
  r.v[0]=x0.x; r.v[1]=x0.y; r.v[2]=x0.z; r.v[3]=x0.w;
  r.v[4]=x1.x; r.v[5]=x1.y; r.v[6]=x1.z; r.v[7]=x1.w;
  r.v[8]=x2.x; r.v[9]=x2.y; r.v[10]=x2.z; r.v[11]=x2.w;
  r.v[12]=x3.x; r.v[13]=x3.y; r.v[14]=x3.z; r.v[15]=x3.w;
  return r;
}
__device__ __forceinline__ v16bf cvt16(const f16run& r) {
  v16bf a;
#pragma unroll
  for (int i = 0; i < 16; ++i) a[i] = (__bf16)r.v[i];
  return a;
}
// A-fragment from an fp32 LDS row (row = tile_base + m*ld)
__device__ __forceinline__ v16bf a_frag_f32(const float* row, int k0, int half) {
  return cvt16(load_run16(row + k0 + 8 * half));
}
// B-fragment from TRANSPOSED bf16 weights in LDS: Wt[n*ldk + k]
// lane col n=lane&15; elems 0..7 -> K=k0+8*half+i, elems 8..15 -> K=k0+16+8*half+(i-8) (two 8-runs, 16B aligned)
__device__ __forceinline__ v16bf b_frag_t(const __bf16* Wt, int ldk, int n, int k0, int half) {
  typedef __attribute__((ext_vector_type(8))) __bf16 v8bf;
  const __bf16* p = Wt + n * ldk + k0 + 8 * half;
  v8bf lo = *(const v8bf*)(p);
  v8bf hi = *(const v8bf*)(p + 16);
  return __builtin_shufflevector(lo, hi, 0, 1, 2, 3, 4, 5, 6, 7, 8, 9, 10, 11, 12, 13, 14, 15);
}

// ---- utility kernels ----
__global__ void k_zero4(float4* p, int n4) {
  int i = blockIdx.x * blockDim.x + threadIdx.x;
  if (i < n4) p[i] = make_float4(0.f, 0.f, 0.f, 0.f);
}

__global__ void k_deg(const int* __restrict__ ms, const int* __restrict__ md,
                      const int* __restrict__ rs, const int* __restrict__ rd,
                      float* dom, float* dim, float* dor, float* dir_, int E) {
  int e = blockIdx.x * blockDim.x + threadIdx.x;
  if (e < E) {
    atomicAdd(&dom[ms[e]], 1.f);
    atomicAdd(&dim[md[e]], 1.f);
    atomicAdd(&dor[rs[e]], 1.f);
    atomicAdd(&dir_[rd[e]], 1.f);
  }
}

__global__ void k_rsqrt(float* p, int n) {
  int i = blockIdx.x * blockDim.x + threadIdx.x;
  if (i < n) p[i] = rsqrtf(fmaxf(p[i], 1.f));
}

// out[n][c] = (bn? x*sc[c]+sh[c] : x) * (rowscale? rowscale[n] : 1), float4 over 64-wide rows
__global__ void k_rows4(float4* __restrict__ out, const float4* __restrict__ x,
                        const float* __restrict__ sc, const float* __restrict__ sh,
                        const float* __restrict__ rowscale, int total4) {
  int i = blockIdx.x * blockDim.x + threadIdx.x;
  if (i < total4) {
    int n = i >> 4, c4 = (i & 15) << 2;
    float4 v = x[i];
    if (sc) {
      v.x = v.x * sc[c4 + 0] + sh[c4 + 0];
      v.y = v.y * sc[c4 + 1] + sh[c4 + 1];
      v.z = v.z * sc[c4 + 2] + sh[c4 + 2];
      v.w = v.w * sc[c4 + 3] + sh[c4 + 3];
    }
    if (rowscale) {
      float r = rowscale[n];
      v.x *= r; v.y *= r; v.z *= r; v.w *= r;
    }
    out[i] = v;
  }
}

// edge scatter: 8 lanes per edge, 8 floats each, fp32 atomics (L2-resident accumulator)
__global__ void k_scatter(const float* __restrict__ hs, const int* __restrict__ src,
                          const int* __restrict__ dst, float* __restrict__ agg, int E) {
  int t = blockIdx.x * blockDim.x + threadIdx.x;
  int e = t >> 3;
  if (e < E) {
    int seg = (t & 7) * 8;
    int s = src[e], d = dst[e];
    const float4* p = (const float4*)(hs + (size_t)s * 64 + seg);
    float4 a = p[0], b = p[1];
    float* q = agg + (size_t)d * 64 + seg;
    atomicAdd(q + 0, a.x); atomicAdd(q + 1, a.y); atomicAdd(q + 2, a.z); atomicAdd(q + 3, a.w);
    atomicAdd(q + 4, b.x); atomicAdd(q + 5, b.y); atomicAdd(q + 6, b.z); atomicAdd(q + 7, b.w);
  }
}

// t = ELU((agg * nd[:,None]) @ W + b), plus per-column sum/sumsq for BN
__global__ void __launch_bounds__(128)
k_layer(const float* __restrict__ agg, const float* __restrict__ nd,
        const float* __restrict__ W, const float* __restrict__ bias,
        float* __restrict__ t, float* __restrict__ gsum, float* __restrict__ gsq) {
  __shared__ __attribute__((aligned(32))) __bf16 Wt[64 * 64];   // transposed: Wt[n*64+k]
  __shared__ float bs[64];
  __shared__ __attribute__((aligned(32))) float As[64 * 64];
  __shared__ float csum[64], csq[64];
  int tid = threadIdx.x, w = tid >> 5, lane = tid & 31;
  int half = lane >> 4, m = lane & 15, col = lane & 15;

  for (int i = tid; i < 4096; i += 128) Wt[(i & 63) * 64 + (i >> 6)] = (__bf16)W[i];
  if (tid < 64) { bs[tid] = bias[tid]; csum[tid] = 0.f; csq[tid] = 0.f; }

  int r0 = blockIdx.x * 64 + w * 16;
  float* Aw = As + w * 1024;
  {
    float4* Aw4 = (float4*)Aw;
    for (int idx = lane; idx < 256; idx += 32) {          // 16 rows x 16 float4
      int r = idx >> 4, c4 = idx & 15, row = r0 + r;
      float4 v = make_float4(0.f, 0.f, 0.f, 0.f);
      if (row < NNODES) {
        v = ((const float4*)(agg + (size_t)row * 64))[c4];
        float s = nd[row];
        v.x *= s; v.y *= s; v.z *= s; v.w *= s;
      }
      Aw4[idx] = v;
    }
  }
  __syncthreads();

  const float* Arow = Aw + m * 64;
  v16bf a0 = a_frag_f32(Arow, 0, half);
  v16bf a1 = a_frag_f32(Arow, 32, half);
  bool full = (r0 + 16) <= NNODES;
#pragma unroll
  for (int nt = 0; nt < 4; ++nt) {
    v8f acc = {};
    acc = WMMA_BF16(a0, b_frag_t(Wt, 64, nt * 16 + col, 0, half), acc);
    acc = WMMA_BF16(a1, b_frag_t(Wt, 64, nt * 16 + col, 32, half), acc);
    int cc = nt * 16 + col;
    float bb = bs[cc], ls = 0.f, lq = 0.f;
    if (full) {
      float* tp = t + (size_t)(r0 + 8 * half) * 64 + cc;
#pragma unroll
      for (int v = 0; v < 8; ++v) {
        float y = acc[v] + bb;
        y = (y > 0.f) ? y : (__expf(y) - 1.f);  // ELU alpha=1
        tp[(size_t)v * 64] = y; ls += y; lq += y * y;
      }
    } else {
#pragma unroll
      for (int v = 0; v < 8; ++v) {
        int row = r0 + v + 8 * half;
        float y = acc[v] + bb;
        y = (y > 0.f) ? y : (__expf(y) - 1.f);
        if (row < NNODES) { t[(size_t)row * 64 + cc] = y; ls += y; lq += y * y; }
      }
    }
    atomicAdd(&csum[cc], ls);
    atomicAdd(&csq[cc], lq);
  }
  __syncthreads();
  if (tid < 64) { atomicAdd(&gsum[tid], csum[tid]); atomicAdd(&gsq[tid], csq[tid]); }
}

__global__ void k_bnp(const float* s, const float* sq, const float* g, const float* bt,
                      float* sc, float* sh, float invN) {
  int c = threadIdx.x;
  float mu = s[c] * invN;
  float var = sq[c] * invN - mu * mu;
  float scale = g[c] * rsqrtf(var + 1e-5f);
  sc[c] = scale;
  sh[c] = bt[c] - mu * scale;
}

// fully fused gating head: cat@gW1 + PReLU -> @gW2 + sigmoid -> fuse -> @fcW + fcb
__global__ void __launch_bounds__(128)
k_fuse(const float* __restrict__ h1, const float* __restrict__ h2,
       const float* __restrict__ gW1, const float* __restrict__ gb1,
       const float* __restrict__ prelu_a,
       const float* __restrict__ gW2, const float* __restrict__ gb2,
       const float* __restrict__ fcW, const float* __restrict__ fcb,
       float* __restrict__ out) {
  extern __shared__ __attribute__((aligned(32))) char smem[];
  __bf16* W1t = (__bf16*)smem;            // transposed 64n x 256k
  __bf16* W2t = W1t + 256 * 64;           // transposed 64n x 64k
  __bf16* Wft = W2t + 64 * 64;            // transposed 144n x 64k (129 padded)
  float*  b1s = (float*)(Wft + 144 * 64);
  float*  b2s = b1s + 64;
  float*  bfs = b2s + 64;                 // 144
  float*  H12 = bfs + 144;                // 4 waves x 16 x 128
  float*  Zs  = H12 + 4 * 16 * 128;       // 4 waves x 16 x 64

  int tid = threadIdx.x, w = tid >> 5, lane = tid & 31;
  int half = lane >> 4, m = lane & 15, col = lane & 15;

  for (int i = tid; i < 256 * 64; i += 128) W1t[(i & 63) * 256 + (i >> 6)] = (__bf16)gW1[i];
  for (int i = tid; i < 64 * 64;  i += 128) W2t[(i & 63) * 64 + (i >> 6)] = (__bf16)gW2[i];
  for (int i = tid; i < 144 * 64; i += 128) {
    int n = i >> 6, k = i & 63;
    Wft[i] = (__bf16)((n < 129) ? fcW[k * 129 + n] : 0.f);
  }
  for (int i = tid; i < 64;  i += 128) { b1s[i] = gb1[i]; b2s[i] = gb2[i]; }
  for (int i = tid; i < 144; i += 128) bfs[i] = (i < 129) ? fcb[i] : 0.f;

  int r0 = blockIdx.x * 64 + w * 16;
  float* Hw = H12 + w * 16 * 128;
  float* Zw = Zs + w * 16 * 64;
  {
    float4* Hw4 = (float4*)Hw;
    for (int idx = lane; idx < 512; idx += 32) {          // 16 rows x 32 float4
      int r = idx >> 5, c4i = idx & 31, c = c4i << 2, row = r0 + r;
      float4 v = make_float4(0.f, 0.f, 0.f, 0.f);
      if (row < NNODES)
        v = (c < 64) ? ((const float4*)(h1 + (size_t)row * 64))[c4i]
                     : ((const float4*)(h2 + (size_t)row * 64))[c4i - 16];
      Hw4[idx] = v;
    }
  }
  __syncthreads();

  float alpha = prelu_a[0];
  bool full = (r0 + 16) <= NNODES;

  // GEMM1: cat(16x256) @ gW1(256x64); cat A-fragments built from float4 LDS runs.
  // Each K-block of 32 lies wholly in one cat segment (seg = kb>>1).
  // |h1-h2| via fmaxf(d,-d) form: fabsf over vectorized LDS loads trips a
  // gfx1250 ISel bug (i32<->v2f32 sign-mask bitcast).
  v16bf afr[8];
#pragma unroll
  for (int kb = 0; kb < 8; ++kb) {
    int seg = kb >> 1;
    int base = (kb & 1) * 32 + 8 * half;
    f16run r1 = load_run16(Hw + m * 128 + base);
    f16run r2 = load_run16(Hw + m * 128 + 64 + base);
    f16run cv;
#pragma unroll
    for (int i = 0; i < 16; ++i) {
      float v1 = r1.v[i], v2 = r2.v[i];
      cv.v[i] = (seg == 0) ? v1
              : (seg == 1) ? v2
              : (seg == 2) ? fmaxf(v1 - v2, v2 - v1)
                           : v1 * v2;
    }
    afr[kb] = cvt16(cv);
  }
#pragma unroll
  for (int nt = 0; nt < 4; ++nt) {
    v8f acc = {};
#pragma unroll
    for (int kb = 0; kb < 8; ++kb)
      acc = WMMA_BF16(afr[kb], b_frag_t(W1t, 256, nt * 16 + col, kb * 32, half), acc);
    int cc = nt * 16 + col;
    float bb = b1s[cc];
#pragma unroll
    for (int v = 0; v < 8; ++v) {
      float z = acc[v] + bb;
      z = (z > 0.f) ? z : alpha * z;  // PReLU
      Zw[(v + 8 * half) * 64 + cc] = z;
    }
  }
  __syncthreads();

  // GEMM2: z @ gW2 -> sigmoid gate -> fused = g*h1 + (1-g)*h2
  {
    const float* Zrow = Zw + m * 64;
    v16bf a0 = a_frag_f32(Zrow, 0, half);
    v16bf a1 = a_frag_f32(Zrow, 32, half);
    float fus[4][8];
#pragma unroll
    for (int nt = 0; nt < 4; ++nt) {
      v8f acc = {};
      acc = WMMA_BF16(a0, b_frag_t(W2t, 64, nt * 16 + col, 0, half), acc);
      acc = WMMA_BF16(a1, b_frag_t(W2t, 64, nt * 16 + col, 32, half), acc);
      int cc = nt * 16 + col;
      float bb = b2s[cc];
#pragma unroll
      for (int v = 0; v < 8; ++v) {
        int mm = v + 8 * half;
        float g = 1.f / (1.f + __expf(-(acc[v] + bb)));
        float v1 = Hw[mm * 128 + cc], v2 = Hw[mm * 128 + 64 + cc];
        fus[nt][v] = g * v1 + (1.f - g) * v2;
      }
    }
    __syncthreads();
#pragma unroll
    for (int nt = 0; nt < 4; ++nt) {
      int cc = nt * 16 + col;
#pragma unroll
      for (int v = 0; v < 8; ++v) Zw[(v + 8 * half) * 64 + cc] = fus[nt][v];
    }
    __syncthreads();
  }

  // GEMM3: fused(16x64) @ fcW(64x129, padded 144) + fcb
  {
    const float* Zrow = Zw + m * 64;
    v16bf a0 = a_frag_f32(Zrow, 0, half);
    v16bf a1 = a_frag_f32(Zrow, 32, half);
#pragma unroll
    for (int nt = 0; nt < 9; ++nt) {
      v8f acc = {};
      acc = WMMA_BF16(a0, b_frag_t(Wft, 64, nt * 16 + col, 0, half), acc);
      acc = WMMA_BF16(a1, b_frag_t(Wft, 64, nt * 16 + col, 32, half), acc);
      int cc = nt * 16 + col;
      float bb = bfs[cc];
      bool colok = (nt < 8) || (cc < 129);  // folds to true for nt<8 at unroll
      if (full) {
        if (colok) {
          float* op = out + (size_t)(r0 + 8 * half) * 129 + cc;
#pragma unroll
          for (int v = 0; v < 8; ++v) op[(size_t)v * 129] = acc[v] + bb;
        }
      } else {
#pragma unroll
        for (int v = 0; v < 8; ++v) {
          int row = r0 + v + 8 * half;
          if (row < NNODES && colok) out[(size_t)row * 129 + cc] = acc[v] + bb;
        }
      }
    }
  }
}

extern "C" void kernel_launch(void* const* d_in, const int* in_sizes, int n_in,
                              void* d_out, int out_size, void* d_ws, size_t ws_size,
                              hipStream_t stream) {
  const float* x    = (const float*)d_in[0];
  const int* msrc   = (const int*)d_in[1];
  const int* mdst   = (const int*)d_in[2];
  const int* rsrc   = (const int*)d_in[3];
  const int* rdst   = (const int*)d_in[4];
  const float* soc_W1 = (const float*)d_in[5],  *soc_b1 = (const float*)d_in[6];
  const float* soc_g1 = (const float*)d_in[7],  *soc_bt1 = (const float*)d_in[8];
  const float* soc_W2 = (const float*)d_in[9],  *soc_b2 = (const float*)d_in[10];
  const float* soc_g2 = (const float*)d_in[11], *soc_bt2 = (const float*)d_in[12];
  const float* rt_W1  = (const float*)d_in[13], *rt_b1 = (const float*)d_in[14];
  const float* rt_g1  = (const float*)d_in[15], *rt_bt1 = (const float*)d_in[16];
  const float* rt_W2  = (const float*)d_in[17], *rt_b2 = (const float*)d_in[18];
  const float* rt_g2  = (const float*)d_in[19], *rt_bt2 = (const float*)d_in[20];
  const float* gW1 = (const float*)d_in[21], *gb1 = (const float*)d_in[22];
  const float* prelu_a = (const float*)d_in[23];
  const float* gW2 = (const float*)d_in[24], *gb2 = (const float*)d_in[25];
  const float* fcW = (const float*)d_in[26], *fcb = (const float*)d_in[27];
  float* out = (float*)d_out;

  float* ws  = (float*)d_ws;
  float* nsm = ws;
  float* ndm = ws + (size_t)NNODES;
  float* nsr = ws + 2 * (size_t)NNODES;
  float* ndr = ws + 3 * (size_t)NNODES;
  float* hs  = ws + 4 * (size_t)NNODES;
  float* agg = hs  + 64 * (size_t)NNODES;
  float* tb  = agg + 64 * (size_t)NNODES;
  float* h1  = tb  + 64 * (size_t)NNODES;
  float* h2  = h1  + 64 * (size_t)NNODES;
  float* stats = h2 + 64 * (size_t)NNODES;  // 128 floats
  float* bnsc  = stats + 128;               // 64
  float* bnsh  = bnsc + 64;                 // 64

  auto zero = [&](float* p, size_t n) {    // n multiple of 4
    int n4 = (int)(n >> 2);
    k_zero4<<<(n4 + 255) / 256, 256, 0, stream>>>((float4*)p, n4);
  };

  const int nb_rows4 = (NNODES * 16 + 255) / 256;
  const int nb_gemm  = (NNODES + 63) / 64;
  const int nb_scat  = (int)(((size_t)NEDGES * 8 + 255) / 256);
  const float invN = 1.0f / (float)NNODES;

  // degrees -> rsqrt norms
  zero(ws, 4 * (size_t)NNODES);
  k_deg<<<(NEDGES + 255) / 256, 256, 0, stream>>>(msrc, mdst, rsrc, rdst, nsm, ndm, nsr, ndr, NEDGES);
  k_rsqrt<<<(4 * NNODES + 255) / 256, 256, 0, stream>>>(ws, 4 * NNODES);

  auto layer = [&](const float* xin, const float* sc, const float* sh,
                   const float* ns, const float* nd, const int* src, const int* dst,
                   const float* W, const float* b, const float* g, const float* bt) {
    k_rows4<<<nb_rows4, 256, 0, stream>>>((float4*)hs, (const float4*)xin, sc, sh, ns, NNODES * 16);
    zero(agg, 64 * (size_t)NNODES);
    k_scatter<<<nb_scat, 256, 0, stream>>>(hs, src, dst, agg, NEDGES);
    zero(stats, 128);
    k_layer<<<nb_gemm, 128, 0, stream>>>(agg, nd, W, b, tb, stats, stats + 64);
    k_bnp<<<1, 64, 0, stream>>>(stats, stats + 64, g, bt, bnsc, bnsh, invN);
  };

  // branch 1: social graph
  layer(x,  nullptr, nullptr, nsm, ndm, msrc, mdst, soc_W1, soc_b1, soc_g1, soc_bt1);
  layer(tb, bnsc, bnsh,       nsm, ndm, msrc, mdst, soc_W2, soc_b2, soc_g2, soc_bt2);
  k_rows4<<<nb_rows4, 256, 0, stream>>>((float4*)h1, (const float4*)tb, bnsc, bnsh, nullptr, NNODES * 16);

  // branch 2: retweet graph
  layer(x,  nullptr, nullptr, nsr, ndr, rsrc, rdst, rt_W1, rt_b1, rt_g1, rt_bt1);
  layer(tb, bnsc, bnsh,       nsr, ndr, rsrc, rdst, rt_W2, rt_b2, rt_g2, rt_bt2);
  k_rows4<<<nb_rows4, 256, 0, stream>>>((float4*)h2, (const float4*)tb, bnsc, bnsh, nullptr, NNODES * 16);

  // fused gating head + classifier
  size_t smem = sizeof(__bf16) * (256 * 64 + 64 * 64 + 144 * 64)
              + sizeof(float) * (64 + 64 + 144)
              + sizeof(float) * (4 * 16 * 128 + 4 * 16 * 64);
  k_fuse<<<nb_gemm, 128, smem, stream>>>(h1, h2, gW1, gb1, prelu_a, gW2, gb2, fcW, fcb, out);
}
